// MemoryCritic_33517924778507
// MI455X (gfx1250) — matrix operand order repaired and checked by reading, not compiled, not run on hardware
//
#include <hip/hip_runtime.h>

// ---------------------------------------------------------------------------
// MemoryCritic kNN kernel for MI455X (gfx1250, wave32, WMMA)
//   obs      [256, 32]  f32
//   action   [256, 2]   f32
//   mem_sa   [500000,34] f32
//   mem_Q    [500000,1] f32
//   out      [256]      f32
//
// score(b,n) = ||m_n||^2 - 2*q_b.m_n   (||q||^2 dropped: softmax/top-k invariant)
// Dot products via V_WMMA_F32_16X16X32_F16 (K=32 covers obs exactly; 2nd
// zero-padded WMMA covers the 2 action dims).
// ---------------------------------------------------------------------------

typedef __attribute__((ext_vector_type(16))) _Float16 v16h;
typedef __attribute__((ext_vector_type(8)))  float    v8f;

#define KNN   10
#define NMEM  500000
#define DDIM  34
#define OBSD  32
#define QTILES 16              // 256 queries / 16
#define SLICES 50
#define TILES_TOTAL (NMEM / 16)              // 31250 (exact)
#define TILES_PER_SLICE (TILES_TOTAL / SLICES) // 625 (exact)
#define WAVES 4                 // 128-thread block = 4 wave32
#define FINF  3.402823466e38f

__device__ __forceinline__ void topk_insert(float (&bs)[KNN], int (&bi)[KNN],
                                            float s, int ix) {
  // bs[] kept sorted ascending; fast-reject against current worst.
  if (s < bs[KNN - 1]) {
    float cs = s; int ci = ix;
#pragma unroll
    for (int j = 0; j < KNN; ++j) {
      if (cs < bs[j]) {
        float tf = bs[j]; bs[j] = cs; cs = tf;
        int   ti = bi[j]; bi[j] = ci; ci = ti;
      }
    }
  }
}

// ------------------------------ Phase 1 ------------------------------------
// grid (QTILES, SLICES), block 128. Each wave streams 16-row N-tiles of its
// slice, does 2 WMMAs vs its 16-query A tile, and maintains per-lane top-10.
__global__ __launch_bounds__(128) void knn_phase1(
    const float* __restrict__ obs, const float* __restrict__ act,
    const float* __restrict__ mem_sa,
    float* __restrict__ wsS, int* __restrict__ wsI)
{
  const int qt    = blockIdx.x;       // query tile (16 rows)
  const int slice = blockIdx.y;
  const int tid   = threadIdx.x;
  const int lane  = tid & 31;
  const int wid   = tid >> 5;
  const int hi    = (lane >> 4) & 1;  // lane half (K-range selector)
  const int m     = lane & 15;        // A row / D column owned by this lane

  // ---- A matrix (16x32 f16): lane L<16 holds K 0..15; L>=16 K 16..31
  const float* qobs = obs + (size_t)(qt * 16 + m) * OBSD;
  v16h a0;
#pragma unroll
  for (int h = 0; h < 16; ++h) {
    int k = h + hi * 16;
    a0[h] = (_Float16)qobs[k];
  }
  v16h a1;
#pragma unroll
  for (int h = 0; h < 16; ++h) a1[h] = (_Float16)0.0f;
  if (hi == 0) {                       // global dims 32,33 = action
    a1[0] = (_Float16)act[(qt * 16 + m) * 2 + 0];
    a1[1] = (_Float16)act[(qt * 16 + m) * 2 + 1];
  }

  float bs[KNN]; int bi[KNN];
#pragma unroll
  for (int j = 0; j < KNN; ++j) { bs[j] = FINF; bi[j] = 0; }

  __shared__ float sTile[WAVES][256];            // per-wave 16x16 score tile
  __shared__ float mS[16][2 * WAVES * KNN];      // block-level merge buffers
  __shared__ int   mI[16][2 * WAVES * KNN];

  const int t0 = slice * TILES_PER_SLICE;
  const int t1 = t0 + TILES_PER_SLICE;
  const int n  = lane & 15;            // B column = memory row within tile

  for (int t = t0 + wid; t < t1; t += WAVES) {
    const float* mrow = mem_sa + (size_t)(t * 16 + n) * DDIM;
    __builtin_prefetch(mem_sa + (size_t)((t + WAVES) * 16 + n) * DDIM, 0, 1);

    // ---- B matrix (32x16 f16): lanes 0-15 hold K 0..15, lanes 16-31 K 16..31
    float ssq = 0.0f;
    v16h b0;
    const int kb = hi * 16;
#pragma unroll
    for (int h = 0; h < 16; ++h) {
      float v = mrow[kb + h];
      b0[h] = (_Float16)v;
      ssq += v * v;
    }
    v16h b1;
#pragma unroll
    for (int h = 0; h < 16; ++h) b1[h] = (_Float16)0.0f;
    if (hi == 0) {                     // K-chunk2: lanes<16 hold dims 32..47
      float v0 = mrow[32], v1 = mrow[33];
      b1[0] = (_Float16)v0; b1[1] = (_Float16)v1;
      ssq += v0 * v0 + v1 * v1;
    }
    // row norm: combine the two lane halves
    float nrm = ssq + __shfl_xor(ssq, 16, 32);

    v8f c;
#pragma unroll
    for (int r = 0; r < 8; ++r) c[r] = 0.0f;
    c = __builtin_amdgcn_wmma_f32_16x16x32_f16(false, a0, false, b0,
                                               (short)0, c, false, false);
    c = __builtin_amdgcn_wmma_f32_16x16x32_f16(false, a1, false, b1,
                                               (short)0, c, false, false);

    // D layout: vgpr r -> query row (r + hi*8), lane&15 -> column n.
    // score = ||m||^2 - 2*dot ; stash query-row-major in this wave's LDS tile.
#pragma unroll
    for (int r = 0; r < 8; ++r) {
      int rowm = r + hi * 8;
      sTile[wid][rowm * 16 + n] = nrm - 2.0f * c[r];
    }
    asm volatile("s_wait_dscnt 0" ::: "memory");  // wave-local LDS RAW

    // rescan: this lane owns query row m, half the columns
    const int base = t * 16;
#pragma unroll
    for (int cc = 0; cc < 8; ++cc) {
      int col = hi * 8 + cc;
      float s = sTile[wid][m * 16 + col];
      topk_insert(bs, bi, s, base + col);
    }
  }

  // ---- block merge: 2*WAVES contributor lists of 10 per query row
  const int slot = wid * 2 + hi;
#pragma unroll
  for (int j = 0; j < KNN; ++j) {
    mS[m][slot * KNN + j] = bs[j];
    mI[m][slot * KNN + j] = bi[j];
  }
  __syncthreads();

  if (tid < 16) {
    float fb[KNN]; int fi[KNN];
#pragma unroll
    for (int j = 0; j < KNN; ++j) { fb[j] = FINF; fi[j] = 0; }
    for (int c2 = 0; c2 < 2 * WAVES * KNN; ++c2)
      topk_insert(fb, fi, mS[tid][c2], mI[tid][c2]);
    float* oS = wsS + ((size_t)(qt * SLICES + slice) * 16 + tid) * KNN;
    int*   oI = wsI + ((size_t)(qt * SLICES + slice) * 16 + tid) * KNN;
#pragma unroll
    for (int j = 0; j < KNN; ++j) { oS[j] = fb[j]; oI[j] = fi[j]; }
  }
}

// ------------------------------ Phase 2 ------------------------------------
// one wave per query: merge SLICES*KNN candidates, recompute exact f32
// distances for the winners, softmax(raw dists), gather Q, weighted sum.
__global__ __launch_bounds__(32) void knn_phase2(
    const float* __restrict__ wsS, const int* __restrict__ wsI,
    const float* __restrict__ obs, const float* __restrict__ act,
    const float* __restrict__ mem_sa, const float* __restrict__ mem_Q,
    float* __restrict__ out)
{
  const int b    = blockIdx.x;
  const int qt   = b >> 4, row = b & 15;
  const int lane = threadIdx.x;

  float bs[KNN]; int bi[KNN];
#pragma unroll
  for (int j = 0; j < KNN; ++j) { bs[j] = FINF; bi[j] = 0; }

  const int total = SLICES * KNN;   // 500 candidates
  for (int c = lane; c < total; c += 32) {
    int s = c / KNN, j = c % KNN;
    size_t off = ((size_t)(qt * SLICES + s) * 16 + row) * KNN + j;
    topk_insert(bs, bi, wsS[off], wsI[off]);
  }

  __shared__ float fS[32 * KNN];
  __shared__ int   fI[32 * KNN];
#pragma unroll
  for (int j = 0; j < KNN; ++j) {
    fS[lane * KNN + j] = bs[j];
    fI[lane * KNN + j] = bi[j];
  }
  __syncthreads();

  if (lane == 0) {
    float gb[KNN]; int gi[KNN];
#pragma unroll
    for (int j = 0; j < KNN; ++j) { gb[j] = FINF; gi[j] = 0; }
    for (int c = 0; c < 32 * KNN; ++c) topk_insert(gb, gi, fS[c], fI[c]);

    // exact f32 recompute of the 10 winning distances (f16 WMMA selected them)
    const float* qo = obs + (size_t)b * OBSD;
    const float* qa = act + (size_t)b * 2;
#pragma unroll
    for (int j = 0; j < KNN; ++j) {
      const float* mr = mem_sa + (size_t)gi[j] * DDIM;
      float dot = 0.0f, nn = 0.0f;
      for (int k = 0; k < OBSD; ++k) { dot += qo[k] * mr[k]; nn += mr[k] * mr[k]; }
      dot += qa[0] * mr[32] + qa[1] * mr[33];
      nn  += mr[32] * mr[32] + mr[33] * mr[33];
      gb[j] = nn - 2.0f * dot;
    }
    float mx = gb[0];
#pragma unroll
    for (int j = 1; j < KNN; ++j) mx = fmaxf(mx, gb[j]);
    float wsum = 0.0f, acc = 0.0f;
#pragma unroll
    for (int j = 0; j < KNN; ++j) {
      float w = __expf(gb[j] - mx);   // softmax over raw (shifted) distances
      wsum += w;
      acc  += w * mem_Q[gi[j]];
    }
    out[b] = acc / wsum;
  }
}

// ------------------------------ Launcher -----------------------------------
extern "C" void kernel_launch(void* const* d_in, const int* in_sizes, int n_in,
                              void* d_out, int out_size, void* d_ws, size_t ws_size,
                              hipStream_t stream) {
  (void)in_sizes; (void)n_in; (void)out_size; (void)ws_size;
  const float* obs    = (const float*)d_in[0];
  const float* act    = (const float*)d_in[1];
  const float* mem_sa = (const float*)d_in[2];
  const float* mem_Q  = (const float*)d_in[3];
  float* out = (float*)d_out;

  const size_t cand = (size_t)QTILES * SLICES * 16 * KNN;  // 128000
  float* wsS = (float*)d_ws;
  int*   wsI = (int*)((char*)d_ws + cand * sizeof(float)); // ~1 MB total

  dim3 g1(QTILES, SLICES);
  knn_phase1<<<g1, 128, 0, stream>>>(obs, act, mem_sa, wsS, wsI);
  knn_phase2<<<256, 32, 0, stream>>>(wsS, wsI, obs, act, mem_sa, mem_Q, out);
}
